// Llama4TextDecoderLayer_37151467110426
// MI455X (gfx1250) — matrix operand order repaired and checked by reading, not compiled, not run on hardware
//
#include <hip/hip_runtime.h>

// ---------------------------------------------------------------------------
// Llama4 decoder layer for MI455X (gfx1250, wave32).
//  - All matmuls: v_wmma_f32_16x16x32_bf16 (16x16x32 tiles).
//  - Activations stored bf16 in HBM; staged into LDS with
//    global_load_async_to_lds_b128 (ASYNCcnt) -- no VGPR round trip.
//  - Weights stream as f32 from HBM (dominant traffic, ~0.5 GB -> ~21 us at
//    23.3 TB/s) and are converted to bf16 at LDS staging; next K-tile is
//    prefetched with global_prefetch_b8.
//  - Top-1 MoE routing done sparsely (8x less expert weight traffic than the
//    dense reference einsum; numerically identical since gates are 0).
// Workspace usage: ~200 MB from d_ws (bump allocated).
// ---------------------------------------------------------------------------

typedef __bf16 bf16;
typedef __attribute__((ext_vector_type(16))) __bf16 v16bf;
typedef __attribute__((ext_vector_type(8)))  float  v8f;

#define DEVI __device__ __forceinline__

constexpr int B_  = 2;
constexpr int S_  = 1024;
constexpr int H_  = 2048;
constexpr int NH  = 16;
constexpr int NKV = 4;
constexpr int HD  = 128;
constexpr int E_  = 8;
constexpr int I_  = 2048;
constexpr int T_  = B_ * S_;
constexpr float SCALE_ = 0.08838834764831845f;  // HD^-0.5
constexpr float NEG_   = -1e30f;

DEVI bf16 f2bf(float x) { return (bf16)x; }

// Async copy of 16 bytes HBM -> LDS (no VGPR data). LDS byte address is the
// low 32 bits of the generic pointer (flat LDS aperture keeps offset in
// addr[31:0]). Tracked by ASYNCcnt.
DEVI void async_b128(unsigned lds, unsigned long long gaddr) {
  asm volatile("global_load_async_to_lds_b128 %0, %1, off"
               :: "v"(lds), "v"(gaddr) : "memory");
}
DEVI void wait_async() { asm volatile("s_wait_asynccnt 0x0" ::: "memory"); }
DEVI unsigned lds_addr(const void* p) { return (unsigned)(uintptr_t)p; }

// --- WMMA fragment loaders (per CDNA5 ISA 7.12.2 VGPR layouts) -------------
// A (16x32 bf16, MxK): lanes 0-15 M=0..15 K groups {0-7,16-23},
//                      lanes 16-31 same M, K groups {8-15,24-31}.
DEVI v16bf frag_a(const bf16* p, int ld, int row0, int k0) {
  int lane = threadIdx.x & 31;
  int g = lane >> 4, m = lane & 15;
  const bf16* r = p + (row0 + m) * ld + k0 + 8 * g;
  v16bf f;
#pragma unroll
  for (int i = 0; i < 4; i++) { f[2*i]   = r[2*i];      f[2*i+1] = r[2*i+1]; }
#pragma unroll
  for (int i = 0; i < 4; i++) { f[8+2*i] = r[16+2*i];   f[9+2*i] = r[16+2*i+1]; }
  return f;
}
// B (32x16 bf16, KxN) staged N-major in LDS (Bt[n][k]): lanes 0-15 K groups
// {0-7,16-23}, lanes 16-31 K groups {8-15,24-31} (matches frag_a).
DEVI v16bf frag_b(const bf16* p, int ld, int n0, int k0) {
  int lane = threadIdx.x & 31;
  int g = lane >> 4, n = lane & 15;
  const bf16* r = p + (n0 + n) * ld + k0 + 8 * g;
  v16bf f;
#pragma unroll
  for (int j = 0; j < 16; j++) f[j] = r[j < 8 ? j : j + 8];
  return f;
}

DEVI v8f wmma_bf16(v16bf a, v16bf b, v8f c) {
  return __builtin_amdgcn_wmma_f32_16x16x32_bf16(false, a, false, b, (short)0,
                                                 c, false, false);
}

// ---------------------------------------------------------------------------
// RMSNorm: one block per token; f32 in, bf16 out.
// ---------------------------------------------------------------------------
__global__ __launch_bounds__(256) void rmsnorm_kernel(const float* __restrict__ x,
                                                      const float* __restrict__ w,
                                                      bf16* __restrict__ y) {
  int t = blockIdx.x;
  const float* xp = x + (size_t)t * H_;
  float ss = 0.f;
  for (int i = threadIdx.x; i < H_; i += 256) { float v = xp[i]; ss += v * v; }
  __shared__ float red[256];
  red[threadIdx.x] = ss;
  __syncthreads();
  for (int o = 128; o > 0; o >>= 1) {
    if (threadIdx.x < o) red[threadIdx.x] += red[threadIdx.x + o];
    __syncthreads();
  }
  float sc = rsqrtf(red[0] / (float)H_ + 1e-5f);
  bf16* yp = y + (size_t)t * H_;
  for (int i = threadIdx.x; i < H_; i += 256) yp[i] = f2bf(xp[i] * sc * w[i]);
}

// ---------------------------------------------------------------------------
// Generic WMMA GEMM:  C[M,N] (f32) = gather(A)[M,K] * B[K,N] (+ residual)
// A: bf16 HBM, staged via async-to-LDS (16B granules, per-lane addresses so
//    expert row-gather is free). B: f32 HBM, converted to bf16 at staging.
// 128x128x64 tiles, 8 waves (2x4), each wave 64x32 via 4x2 accumulators.
// rowmap/segOff != null => expert gather/scatter mode (blockIdx.z = expert).
// ---------------------------------------------------------------------------
__global__ __launch_bounds__(256) void gemm_wmma(
    const bf16* __restrict__ A, int lda,
    const float* __restrict__ Bw, int ldb, long long strideB,
    float* __restrict__ C, int ldc,
    const float* __restrict__ resid,
    int M, int N, int K,
    const int* __restrict__ rowmap, const int* __restrict__ segOff) {
  __shared__ bf16 As[128][72];
  __shared__ bf16 Bs[128][72];

  int e = blockIdx.z;
  const float* Bp = Bw + (size_t)e * strideB;
  int rowBase = 0, rowCount = M;
  if (segOff) { rowBase = segOff[e]; rowCount = segOff[e + 1] - rowBase; }
  int r0 = blockIdx.y * 128;
  if (r0 >= rowCount) return;
  int c0 = blockIdx.x * 128;

  int tid = threadIdx.x, lane = tid & 31, wv = tid >> 5;
  int wm0 = (wv >> 2) * 64, wn0 = (wv & 3) * 32;

  const v8f vz = {0.f, 0.f, 0.f, 0.f, 0.f, 0.f, 0.f, 0.f};
  v8f acc[4][2];
#pragma unroll
  for (int mi = 0; mi < 4; mi++)
#pragma unroll
    for (int ni = 0; ni < 2; ni++) acc[mi][ni] = vz;

  // A staging: thread -> (row = tid/2, 32-elem half = tid&1); 4 x b128 async.
  int arow_l = tid >> 1, acol = (tid & 1) * 32;
  int lr = r0 + arow_l;
  int cl = lr < rowCount ? lr : rowCount - 1;
  int arow_g = rowmap ? rowmap[rowBase + cl] : cl;
  unsigned long long agbase =
      (unsigned long long)(uintptr_t)(A + (size_t)arow_g * lda + acol);
  unsigned aldsbase = lds_addr(&As[arow_l][acol]);

  // B staging: thread -> (k row = tid/32 + i*8, 4-col group = (tid&31)*4).
  int bk = tid >> 5, bc4 = (tid & 31) * 4;

  for (int k0 = 0; k0 < K; k0 += 64) {
    // --- A tile: async HBM(bf16) -> LDS, 64 bytes per thread ---
    unsigned long long ag = agbase + (unsigned long long)(2 * k0);
#pragma unroll
    for (int j = 0; j < 4; j++) async_b128(aldsbase + 16u * j, ag + 16ull * j);
    // --- B tile: f32 load + bf16 convert, N-major scatter into LDS ---
#pragma unroll
    for (int i = 0; i < 8; i++) {
      int kk = bk + i * 8;
      float4 vb = *reinterpret_cast<const float4*>(Bp + (size_t)(k0 + kk) * ldb + c0 + bc4);
      Bs[bc4 + 0][kk] = f2bf(vb.x); Bs[bc4 + 1][kk] = f2bf(vb.y);
      Bs[bc4 + 2][kk] = f2bf(vb.z); Bs[bc4 + 3][kk] = f2bf(vb.w);
    }
    if (k0 + 64 < K)  // prefetch next weight K-tile into cache
      __builtin_prefetch(Bp + (size_t)(k0 + 64 + bk) * ldb + c0 + bc4, 0, 0);
    wait_async();
    __syncthreads();
#pragma unroll
    for (int ks = 0; ks < 64; ks += 32) {
      v16bf af[4], bfv[2];
#pragma unroll
      for (int mi = 0; mi < 4; mi++) af[mi] = frag_a(&As[0][0], 72, wm0 + mi * 16, ks);
#pragma unroll
      for (int ni = 0; ni < 2; ni++) bfv[ni] = frag_b(&Bs[0][0], 72, wn0 + ni * 16, ks);
#pragma unroll
      for (int mi = 0; mi < 4; mi++)
#pragma unroll
        for (int ni = 0; ni < 2; ni++) acc[mi][ni] = wmma_bf16(af[mi], bfv[ni], acc[mi][ni]);
    }
    __syncthreads();
  }

  // Epilogue: C/D layout -> N = lane&15, M = vgpr + 8*(lane>=16).
  int n = lane & 15, half = lane >> 4;
#pragma unroll
  for (int mi = 0; mi < 4; mi++) {
#pragma unroll
    for (int ni = 0; ni < 2; ni++) {
#pragma unroll
      for (int r = 0; r < 8; r++) {
        int lrow = r0 + wm0 + mi * 16 + r + 8 * half;
        if (lrow < rowCount) {
          int grow = rowmap ? rowmap[rowBase + lrow] : lrow;
          int col = c0 + wn0 + ni * 16 + n;
          float v = acc[mi][ni][r];
          if (resid) v += resid[(size_t)grow * ldc + col];
          C[(size_t)grow * ldc + col] = v;
        }
      }
    }
  }
}

// ---------------------------------------------------------------------------
// RoPE + L2-norm (q,k -> bf16) / copy (v -> f32). One wave per (token, head).
// Outputs head-major: q [B,NH,S,HD], k/v [B,NKV,S,HD].
// ---------------------------------------------------------------------------
__global__ __launch_bounds__(256) void rope_l2_kernel(const float* __restrict__ qkv,
                                                      const float* __restrict__ fr,
                                                      bf16* __restrict__ q,
                                                      bf16* __restrict__ k,
                                                      float* __restrict__ v) {
  int gw = blockIdx.x * 8 + (threadIdx.x >> 5);
  int lane = threadIdx.x & 31;
  if (gw >= T_ * (NH + 2 * NKV)) return;
  int t = gw / (NH + 2 * NKV), hh = gw % (NH + 2 * NKV);
  int b = t / S_, s = t % S_;

  if (hh >= NH + NKV) {  // V: plain copy, keep f32
    int vv = hh - NH - NKV;
    const float* src = qkv + (size_t)t * 3072 + (NH + NKV) * HD + vv * HD;
    float* dst = v + (((size_t)(b * NKV + vv)) * S_ + s) * HD;
#pragma unroll
    for (int j = 0; j < 4; j++) { int i = lane + j * 32; dst[i] = src[i]; }
    return;
  }
  const float* src;
  bf16* dst;
  if (hh < NH) {
    src = qkv + (size_t)t * 3072 + hh * HD;
    dst = q + (((size_t)(b * NH + hh)) * S_ + s) * HD;
  } else {
    int kv = hh - NH;
    src = qkv + (size_t)t * 3072 + NH * HD + kv * HD;
    dst = k + (((size_t)(b * NKV + kv)) * S_ + s) * HD;
  }
  float e[4];
  float ss = 0.f;
#pragma unroll
  for (int j = 0; j < 2; j++) {
    int p = lane + j * 32;  // rotary pair index 0..63
    float a = src[2 * p], bb = src[2 * p + 1];
    float c  = fr[((size_t)s * 64 + p) * 2];
    float sn = fr[((size_t)s * 64 + p) * 2 + 1];
    float o0 = a * c - bb * sn, o1 = a * sn + bb * c;
    e[2 * j] = o0; e[2 * j + 1] = o1;
    ss += o0 * o0 + o1 * o1;
  }
#pragma unroll
  for (int o = 16; o >= 1; o >>= 1) ss += __shfl_xor(ss, o, 32);
  float sc = rsqrtf(ss / (float)HD + 1e-6f);
#pragma unroll
  for (int j = 0; j < 2; j++) {
    int p = lane + j * 32;
    dst[2 * p]     = f2bf(e[2 * j]     * sc);
    dst[2 * p + 1] = f2bf(e[2 * j + 1] * sc);
  }
}

// ---------------------------------------------------------------------------
// Flash attention (causal, GQA). One block = 32 query rows of one (b,h).
// Q/K tiles: bf16 in HBM, async-staged to LDS. V: f32, converted+transposed.
// QK^T and P*V via WMMA bf16; online softmax row stats in LDS. bf16 out.
// ---------------------------------------------------------------------------
__global__ __launch_bounds__(256) void flash_attn_kernel(const bf16* __restrict__ q,
                                                         const bf16* __restrict__ k,
                                                         const float* __restrict__ v,
                                                         bf16* __restrict__ out) {
  __shared__ bf16  Qs[32][136];
  __shared__ bf16  Ks[64][136];
  __shared__ bf16  Vts[128][72];
  __shared__ float Ss[32][68];
  __shared__ bf16  Ps[32][72];
  __shared__ float mrow[32], lrow[32], arow[32];

  int tid = threadIdx.x, lane = tid & 31, wv = tid >> 5;
  int q0 = blockIdx.x * 32, h = blockIdx.y, b = blockIdx.z;
  int kvh = h / (NH / NKV);
  const v8f vz = {0.f, 0.f, 0.f, 0.f, 0.f, 0.f, 0.f, 0.f};

  {  // Q tile (32x128 bf16): async HBM -> LDS, 32 bytes per thread
    int row = tid >> 3, cb = (tid & 7) * 16;
    unsigned long long gq = (unsigned long long)(uintptr_t)(
        q + (((size_t)(b * NH + h)) * S_ + q0 + row) * HD + cb);
    unsigned ldsq = lds_addr(&Qs[row][cb]);
    async_b128(ldsq, gq);
    async_b128(ldsq + 16u, gq + 16ull);
  }
  if (tid < 32) { mrow[tid] = -3.0e38f; lrow[tid] = 0.f; }

  v8f oacc[2];
  oacc[0] = vz; oacc[1] = vz;
  int sm0 = (wv >> 2) * 16, sn0 = (wv & 3) * 16;   // S tile 32x64: 2x4 waves
  int pm0 = (wv >> 2) * 16, pn0 = (wv & 3) * 32;   // O tile 32x128: 2x4 waves
  wait_async();
  __syncthreads();

  int krow = tid >> 2, kcb = (tid & 3) * 32;
  unsigned ldsk = lds_addr(&Ks[krow][kcb]);
  int kend = q0 + 32;
  for (int k0 = 0; k0 < kend; k0 += 64) {
    {  // K tile (64x128 bf16): async HBM -> LDS, 64 bytes per thread
      unsigned long long gk = (unsigned long long)(uintptr_t)(
          k + (((size_t)(b * NKV + kvh)) * S_ + k0 + krow) * HD + kcb);
#pragma unroll
      for (int j = 0; j < 4; j++) async_b128(ldsk + 16u * j, gk + 16ull * j);
    }
    {  // V tile (64x128 f32): convert + transpose into LDS
      const float4* gv = reinterpret_cast<const float4*>(
          v + (((size_t)(b * NKV + kvh)) * S_ + k0 + krow) * HD + kcb);
#pragma unroll
      for (int jj = 0; jj < 8; jj++) {
        float4 vvv = gv[jj];
        int c = kcb + 4 * jj;
        Vts[c + 0][krow] = f2bf(vvv.x); Vts[c + 1][krow] = f2bf(vvv.y);
        Vts[c + 2][krow] = f2bf(vvv.z); Vts[c + 3][krow] = f2bf(vvv.w);
      }
    }
    wait_async();
    __syncthreads();

    // S = Q * K^T  (scale + causal mask on store)
    v8f sacc = vz;
#pragma unroll
    for (int kk = 0; kk < 4; kk++) {
      v16bf af  = frag_a(&Qs[0][0], 136, sm0, kk * 32);
      v16bf bfk = frag_b(&Ks[0][0], 136, sn0, kk * 32);
      sacc = wmma_bf16(af, bfk, sacc);
    }
    {
      int n = lane & 15, half = lane >> 4;
#pragma unroll
      for (int r = 0; r < 8; r++) {
        int row = sm0 + r + 8 * half;
        int ki = k0 + sn0 + n;
        float sv = sacc[r] * SCALE_;
        if (ki > q0 + row) sv = NEG_;
        Ss[row][sn0 + n] = sv;
      }
    }
    __syncthreads();

    // Online softmax row update (one thread per query row).
    if (tid < 32) {
      float mo = mrow[tid], bm = mo;
#pragma unroll 8
      for (int j = 0; j < 64; j++) bm = fmaxf(bm, Ss[tid][j]);
      float al = __expf(mo - bm);
      float sum = 0.f;
      for (int j = 0; j < 64; j++) {
        float p = __expf(Ss[tid][j] - bm);
        sum += p;
        Ps[tid][j] = f2bf(p);
      }
      lrow[tid] = lrow[tid] * al + sum;
      mrow[tid] = bm;
      arow[tid] = al;
    }
    __syncthreads();

    // Rescale accumulators and accumulate P * V.
    {
      int half = lane >> 4;
#pragma unroll
      for (int r = 0; r < 8; r++) {
        float al = arow[pm0 + r + 8 * half];
        oacc[0][r] *= al;
        oacc[1][r] *= al;
      }
    }
#pragma unroll
    for (int kk = 0; kk < 2; kk++) {
      v16bf pf = frag_a(&Ps[0][0], 72, pm0, kk * 32);
#pragma unroll
      for (int ni = 0; ni < 2; ni++) {
        v16bf vf = frag_b(&Vts[0][0], 72, pn0 + ni * 16, kk * 32);
        oacc[ni] = wmma_bf16(pf, vf, oacc[ni]);
      }
    }
    __syncthreads();
  }

  {  // finalize: divide by row sums, write token-major bf16 [T, NH*HD]
    int n = lane & 15, half = lane >> 4;
#pragma unroll
    for (int r = 0; r < 8; r++) {
      int row = pm0 + r + 8 * half;
      float inv = 1.0f / lrow[row];
#pragma unroll
      for (int ni = 0; ni < 2; ni++) {
        int col = pn0 + ni * 16 + n;
        out[((size_t)(b * S_ + q0 + row)) * (NH * HD) + h * HD + col] =
            f2bf(oacc[ni][r] * inv);
      }
    }
  }
}

// ---------------------------------------------------------------------------
// Router: one block per token. 8 waves each compute one expert logit; top-1
// argmax + sigmoid gate; write gate-scaled activation (bf16).
// ---------------------------------------------------------------------------
__global__ __launch_bounds__(256) void router_kernel(const bf16* __restrict__ h2,
                                                     const float* __restrict__ wr,
                                                     bf16* __restrict__ h2s,
                                                     int* __restrict__ eid,
                                                     float* __restrict__ gate) {
  int t = blockIdx.x;
  const bf16* base = h2 + (size_t)t * H_;
  int wv = threadIdx.x >> 5, lane = threadIdx.x & 31;
  float acc = 0.f;
#pragma unroll 8
  for (int j = 0; j < H_ / 32; j++) {
    int idx = lane + j * 32;
    acc += (float)base[idx] * wr[(size_t)idx * E_ + wv];
  }
#pragma unroll
  for (int o = 16; o >= 1; o >>= 1) acc += __shfl_xor(acc, o, 32);
  __shared__ float lg[E_];
  __shared__ float bgS;
  if (lane == 0) lg[wv] = acc;
  __syncthreads();
  if (threadIdx.x == 0) {
    float bm = lg[0]; int bi = 0;
    for (int e2 = 1; e2 < E_; e2++) if (lg[e2] > bm) { bm = lg[e2]; bi = e2; }
    bgS = 1.0f / (1.0f + __expf(-bm));
    eid[t] = bi;
    gate[t] = bgS;
  }
  __syncthreads();
  float g = bgS;
  for (int j = 0; j < H_ / 256; j++) {
    int idx = threadIdx.x + j * 256;
    h2s[(size_t)t * H_ + idx] = f2bf((float)base[idx] * g);
  }
}

// ---------------------------------------------------------------------------
// Expert token permutation: counts -> exclusive scan -> scatter. One block.
// ---------------------------------------------------------------------------
__global__ __launch_bounds__(256) void route_scan_kernel(const int* __restrict__ eid,
                                                         int* __restrict__ perm,
                                                         int* __restrict__ segOff) {
  __shared__ int cnt[E_], off[E_ + 1], cur[E_];
  int tid = threadIdx.x;
  if (tid < E_) cnt[tid] = 0;
  __syncthreads();
  for (int t = tid; t < T_; t += 256) atomicAdd(&cnt[eid[t]], 1);
  __syncthreads();
  if (tid == 0) {
    off[0] = 0;
    for (int e2 = 0; e2 < E_; e2++) { off[e2 + 1] = off[e2] + cnt[e2]; cur[e2] = off[e2]; }
  }
  __syncthreads();
  for (int t = tid; t < T_; t += 256) {
    int e2 = eid[t];
    int pos = atomicAdd(&cur[e2], 1);
    perm[pos] = t;
  }
  __syncthreads();
  if (tid <= E_) segOff[tid] = off[tid];
}

// ---------------------------------------------------------------------------
// SwiGLU elementwise: act = u * silu(g) (bf16 out), g = gu[:, :I], u = gu[:, I:].
// ---------------------------------------------------------------------------
__global__ __launch_bounds__(256) void silu_mul_kernel(const float* __restrict__ gu,
                                                       bf16* __restrict__ act) {
  size_t idx = (size_t)blockIdx.x * 256 + threadIdx.x;
  size_t t = idx / I_, i = idx % I_;
  float g = gu[t * (2 * I_) + i];
  float u = gu[t * (2 * I_) + I_ + i];
  act[idx] = f2bf(u * g / (1.0f + __expf(-g)));
}

// ---------------------------------------------------------------------------
// Host-side orchestration.
// ---------------------------------------------------------------------------
extern "C" void kernel_launch(void* const* d_in, const int* in_sizes, int n_in,
                              void* d_out, int out_size, void* d_ws, size_t ws_size,
                              hipStream_t stream) {
  (void)in_sizes; (void)n_in; (void)out_size; (void)ws_size;

  const float* x     = (const float*)d_in[0];   // [B,S,H]
  const float* freqs = (const float*)d_in[1];   // [S,HD/2,2]
  const float* w_ln1 = (const float*)d_in[2];
  const float* w_qkv = (const float*)d_in[3];   // [H,3072]
  const float* w_o   = (const float*)d_in[4];   // [2048,H]
  const float* w_ln2 = (const float*)d_in[5];
  const float* w_rt  = (const float*)d_in[6];   // [H,E]
  const float* w_ge  = (const float*)d_in[7];   // [E,H,2I]
  const float* w_de  = (const float*)d_in[8];   // [E,I,H]
  const float* w_sgu = (const float*)d_in[9];   // [H,2I]
  const float* w_sd  = (const float*)d_in[10];  // [I,H]
  float* out = (float*)d_out;

  char* p = (char*)d_ws;
  auto alloc = [&](size_t bytes) -> char* {
    char* r = p;
    p += (bytes + 255) & ~(size_t)255;
    return r;
  };
  bf16*  h    = (bf16*)alloc((size_t)T_ * H_ * 2);
  float* qkv  = (float*)alloc((size_t)T_ * 3072 * 4);
  bf16*  qb   = (bf16*)alloc((size_t)B_ * NH * S_ * HD * 2);
  bf16*  kb   = (bf16*)alloc((size_t)B_ * NKV * S_ * HD * 2);
  float* vb   = (float*)alloc((size_t)B_ * NKV * S_ * HD * 4);
  bf16*  attn = (bf16*)alloc((size_t)T_ * NH * HD * 2);
  float* x1   = (float*)alloc((size_t)T_ * H_ * 4);
  bf16*  h2   = (bf16*)alloc((size_t)T_ * H_ * 2);
  bf16*  h2s  = (bf16*)alloc((size_t)T_ * H_ * 2);
  float* gus  = (float*)alloc((size_t)T_ * 2 * I_ * 4);
  bf16*  acts = (bf16*)alloc((size_t)T_ * I_ * 2);
  float* sdwn = (float*)alloc((size_t)T_ * H_ * 4);
  float* gur  = (float*)alloc((size_t)T_ * 2 * I_ * 4);
  bf16*  actr = (bf16*)alloc((size_t)T_ * I_ * 2);
  int*   eid  = (int*)alloc((size_t)T_ * 4);
  float* gate = (float*)alloc((size_t)T_ * 4);
  int*   perm = (int*)alloc((size_t)T_ * 4);
  int*   seg  = (int*)alloc(16 * 4);

  // 1. h = rmsnorm(x, w_ln1)           (bf16 out)
  rmsnorm_kernel<<<T_, 256, 0, stream>>>(x, w_ln1, h);
  // 2. qkv = h @ w_qkv
  gemm_wmma<<<dim3(3072 / 128, T_ / 128, 1), 256, 0, stream>>>(
      h, H_, w_qkv, 3072, 0, qkv, 3072, nullptr, T_, 3072, H_, nullptr, nullptr);
  // 3. rope + l2 norm -> q,k (bf16), v (f32), head-major
  rope_l2_kernel<<<(T_ * (NH + 2 * NKV)) / 8, 256, 0, stream>>>(qkv, freqs, qb, kb, vb);
  // 4. flash attention -> attn bf16 [T, NH*HD]
  flash_attn_kernel<<<dim3(S_ / 32, NH, B_), 256, 0, stream>>>(qb, kb, vb, attn);
  // 5. x1 = x + attn @ w_o
  gemm_wmma<<<dim3(H_ / 128, T_ / 128, 1), 256, 0, stream>>>(
      attn, NH * HD, w_o, H_, 0, x1, H_, x, T_, H_, NH * HD, nullptr, nullptr);
  // 6. h2 = rmsnorm(x1, w_ln2)         (bf16 out)
  rmsnorm_kernel<<<T_, 256, 0, stream>>>(x1, w_ln2, h2);
  // 7. router: top-1 expert + sigmoid gate, h2s = h2 * gate (bf16)
  router_kernel<<<T_, 256, 0, stream>>>(h2, w_rt, h2s, eid, gate);
  // 8. expert segments
  route_scan_kernel<<<1, 256, 0, stream>>>(eid, perm, seg);
  // 9. shared expert: gus = h2 @ w_shared_gu
  gemm_wmma<<<dim3(2 * I_ / 128, T_ / 128, 1), 256, 0, stream>>>(
      h2, H_, w_sgu, 2 * I_, 0, gus, 2 * I_, nullptr, T_, 2 * I_, H_, nullptr, nullptr);
  // 10. acts = silu(g) * u  (bf16)
  silu_mul_kernel<<<(T_ * I_) / 256, 256, 0, stream>>>(gus, acts);
  // 11. sdwn = x1 + acts @ w_shared_down
  gemm_wmma<<<dim3(H_ / 128, T_ / 128, 1), 256, 0, stream>>>(
      acts, I_, w_sd, H_, 0, sdwn, H_, x1, T_, H_, I_, nullptr, nullptr);
  // 12. routed gate_up (gather rows per expert, scatter by token)
  gemm_wmma<<<dim3(2 * I_ / 128, T_ / 128, E_), 256, 0, stream>>>(
      h2s, H_, w_ge, 2 * I_, (long long)H_ * 2 * I_, gur, 2 * I_, nullptr,
      T_, 2 * I_, H_, perm, seg);
  // 13. actr = silu(g) * u  (bf16; zero rows stay zero)
  silu_mul_kernel<<<(T_ * I_) / 256, 256, 0, stream>>>(gur, actr);
  // 14. out = sdwn + actr @ w_down_e  (every token covered exactly once)
  gemm_wmma<<<dim3(H_ / 128, T_ / 128, E_), 256, 0, stream>>>(
      actr, I_, w_de, H_, (long long)I_ * H_, out, H_, sdwn,
      T_, H_, I_, perm, seg);
}
